// RefAttention_30588757082735
// MI455X (gfx1250) — matrix operand order repaired and checked by reading, not compiled, not run on hardware
//
#include <hip/hip_runtime.h>
#include <hip/hip_bf16.h>

// ---------------- problem constants ----------------
#define NH     8
#define DHEAD  64
#define HIN    56
#define WOUT   28
#define NIN    3137          // 56*56 + 1
#define NTOK   785           // 28*28 + 1
#define CCH    512
#define BATCH  8
#define MROWS  (BATCH * NTOK)   // 6280
#define KTILES 13               // ceil(785/64)

typedef __attribute__((ext_vector_type(16))) _Float16 v16h;
typedef __attribute__((ext_vector_type(8)))  _Float16 v8h;
typedef __attribute__((ext_vector_type(8)))  float    v8f;
typedef __attribute__((ext_vector_type(4)))  int      v4i;

// ---------------- optional async global->LDS copy path (gfx1250) ----------------
#if defined(__has_builtin)
#  if __has_builtin(__builtin_amdgcn_global_load_async_to_lds_b128) && \
      __has_builtin(__builtin_amdgcn_s_wait_asynccnt)
#    define USE_ASYNC_COPY 1
#  endif
#endif
#ifndef USE_ASYNC_COPY
#  define USE_ASYNC_COPY 0
#endif

#if USE_ASYNC_COPY
typedef __attribute__((address_space(1))) v4i* gv4i_p;   // global v4i*
typedef __attribute__((address_space(3))) v4i* lv4i_p;   // LDS v4i*
__device__ __forceinline__ void async_cp16(void* lds_dst, const void* gsrc) {
  // 16B per active lane, tracked by ASYNCcnt
  __builtin_amdgcn_global_load_async_to_lds_b128(
      (gv4i_p)gsrc, (lv4i_p)lds_dst, 0, 0);
}
__device__ __forceinline__ void async_wait0() {
  __builtin_amdgcn_s_wait_asynccnt(0);
}
#else
__device__ __forceinline__ void async_wait0() {}
#endif

__device__ __forceinline__ v8h zero8() {
  v8h z;
#pragma unroll
  for (int i = 0; i < 8; ++i) z[i] = (_Float16)0.f;
  return z;
}

// ---------------- WMMA helpers (gfx1250 wave32) ----------------
__device__ __forceinline__ v8f wmma_f16(v16h a, v16h b, v8f c) {
  return __builtin_amdgcn_wmma_f32_16x16x32_f16(
      /*neg_a=*/false, a, /*neg_b=*/false, b,
      /*c_mod=*/(short)0, c, /*reuse_a=*/false, /*reuse_b=*/false);
}

// A fragment (16x32, row-major src [m][k], ld halves, ld%8==0).
__device__ __forceinline__ v16h frag_a(const _Float16* p, int ld, int m0, int k0, int lane) {
  const _Float16* r = p + (size_t)(m0 + (lane & 15)) * ld + k0 + ((lane >> 4) << 3);
  v8h lo = *(const v8h*)r;
  v8h hi = *(const v8h*)(r + 16);
  return __builtin_shufflevector(lo, hi, 0,1,2,3,4,5,6,7,8,9,10,11,12,13,14,15);
}

// B fragment with logical B[k][n] = src[n][k] (src: N rows of K, row-major).
__device__ __forceinline__ v16h frag_bT(const _Float16* p, int ld, int n0, int k0, int lane) {
  const _Float16* r = p + (size_t)(n0 + (lane & 15)) * ld + k0 + ((lane >> 4) << 3);
  v8h lo = *(const v8h*)r;
  v8h hi = *(const v8h*)(r + 16);
  return __builtin_shufflevector(lo, hi, 0,1,2,3,4,5,6,7,8,9,10,11,12,13,14,15);
}

// ---------------- kernel 1: f32 -> f16 convert ----------------
__global__ void cvt_f32_f16(const float* __restrict__ s, _Float16* __restrict__ d, int n) {
  int i = blockIdx.x * blockDim.x + threadIdx.x;
  if (i < n) d[i] = (_Float16)s[i];
}

// ---------------- kernel 2: depthwise pool (3x3 s2) + LayerNorm ----------------
__global__ __launch_bounds__(64) void pool_ln_kernel(
    const float* __restrict__ x,       // [B][NIN][512]
    const float* __restrict__ convw,   // [3][3][1][64]
    const float* __restrict__ ln_g, const float* __restrict__ ln_b,
    _Float16* __restrict__ outp)       // [MROWS][512]
{
  int gid = blockIdx.x;
  int tok = gid % NTOK;
  int t2  = gid / NTOK;
  int h   = t2 % NH;
  int b   = t2 / NH;
  int dd  = threadIdx.x;

  float val;
  if (tok == 0) {
    val = x[((size_t)b * NIN) * CCH + h * DHEAD + dd];   // cls passthrough
  } else {
    int oi = tok - 1, oy = oi / WOUT, ox = oi % WOUT;
    float acc = 0.f;
#pragma unroll
    for (int ky = 0; ky < 3; ++ky) {
      int iy = oy * 2 + ky - 1;
      if (iy < 0 || iy >= HIN) continue;
#pragma unroll
      for (int kx = 0; kx < 3; ++kx) {
        int ix = ox * 2 + kx - 1;
        if (ix < 0 || ix >= HIN) continue;
        int t = 1 + iy * HIN + ix;
        acc += x[((size_t)b * NIN + t) * CCH + h * DHEAD + dd] *
               convw[(ky * 3 + kx) * DHEAD + dd];
      }
    }
    val = acc;
  }

  __shared__ float s1[64], s2[64];
  s1[dd] = val; s2[dd] = val * val;
  __syncthreads();
  for (int st = 32; st > 0; st >>= 1) {
    if (dd < st) { s1[dd] += s1[dd + st]; s2[dd] += s2[dd + st]; }
    __syncthreads();
  }
  float mu  = s1[0] * (1.f / 64.f);
  float var = s2[0] * (1.f / 64.f) - mu * mu;
  float y   = (val - mu) * rsqrtf(var + 1e-5f) * ln_g[dd] + ln_b[dd];
  outp[((size_t)b * NTOK + tok) * CCH + h * DHEAD + dd] = (_Float16)y;
}

// ---------------- kernel 3: WMMA GEMM  Y[M x 512] = X[M x 512] @ W^T + bias ----------------
template <int OUTF32>
__global__ __launch_bounds__(256) void gemm512_wmma(
    const _Float16* __restrict__ A,   // [M][512]
    const _Float16* __restrict__ W,   // [512][512] row = out channel
    const float* __restrict__ bias,   // [512]
    _Float16* __restrict__ Yh, float* __restrict__ Yf, int M)
{
  __shared__ alignas(16) _Float16 As[128 * 40];
  __shared__ alignas(16) _Float16 Ws[64 * 40];
  const int tid  = threadIdx.x;
  const int lane = tid & 31, wid = tid >> 5;
  const int wm = wid & 3, wn = wid >> 2;
  const int m_blk = blockIdx.x * 128;
  const int n_blk = blockIdx.y * 64;

  v8f acc[2][2] = {};

  for (int k0 = 0; k0 < 512; k0 += 32) {
    __syncthreads();
#if USE_ASYNC_COPY
    if (m_blk + 128 <= M) {
#pragma unroll
      for (int e = tid; e < 128 * 4; e += 256) {
        int row = e >> 2, seg = (e & 3) << 3;
        async_cp16(&As[row * 40 + seg], &A[(size_t)(m_blk + row) * 512 + k0 + seg]);
      }
    } else {
#pragma unroll
      for (int e = tid; e < 128 * 4; e += 256) {
        int row = e >> 2, seg = (e & 3) << 3;
        int gr = m_blk + row;
        v8h v = zero8();
        if (gr < M) v = *(const v8h*)&A[(size_t)gr * 512 + k0 + seg];
        *(v8h*)&As[row * 40 + seg] = v;
      }
    }
    {
      int row = tid >> 2, seg = (tid & 3) << 3;
      async_cp16(&Ws[row * 40 + seg], &W[(size_t)(n_blk + row) * 512 + k0 + seg]);
    }
    async_wait0();
#else
#pragma unroll
    for (int e = tid; e < 128 * 4; e += 256) {
      int row = e >> 2, seg = (e & 3) << 3;
      int gr = m_blk + row;
      v8h v = zero8();
      if (gr < M) v = *(const v8h*)&A[(size_t)gr * 512 + k0 + seg];
      *(v8h*)&As[row * 40 + seg] = v;
    }
    {
      int row = tid >> 2, seg = (tid & 3) << 3;
      *(v8h*)&Ws[row * 40 + seg] =
          *(const v8h*)&W[(size_t)(n_blk + row) * 512 + k0 + seg];
    }
#endif
    __syncthreads();

    v16h a0 = frag_a(As, 40, wm * 32,      0, lane);
    v16h a1 = frag_a(As, 40, wm * 32 + 16, 0, lane);
    v16h b0 = frag_bT(Ws, 40, wn * 32,      0, lane);
    v16h b1 = frag_bT(Ws, 40, wn * 32 + 16, 0, lane);
    acc[0][0] = wmma_f16(a0, b0, acc[0][0]);
    acc[0][1] = wmma_f16(a0, b1, acc[0][1]);
    acc[1][0] = wmma_f16(a1, b0, acc[1][0]);
    acc[1][1] = wmma_f16(a1, b1, acc[1][1]);
  }

#pragma unroll
  for (int mi = 0; mi < 2; ++mi)
#pragma unroll
    for (int ni = 0; ni < 2; ++ni) {
      int col = n_blk + wn * 32 + ni * 16 + (lane & 15);
      float bv = bias[col];
#pragma unroll
      for (int j = 0; j < 8; ++j) {
        int row = m_blk + wm * 32 + mi * 16 + ((lane >> 4) << 3) + j;
        if (row < M) {
          float v = acc[mi][ni][j] + bv;
          if (OUTF32) Yf[(size_t)row * 512 + col] = v;
          else        Yh[(size_t)row * 512 + col] = (_Float16)v;
        }
      }
    }
}

// ---------------- kernel 4: decomposed rel-pos projections ----------------
__global__ __launch_bounds__(64) void relpos_kernel(
    const _Float16* __restrict__ qlin,      // [MROWS][512]
    const float* __restrict__ rel_h_tab,    // [55][64]
    const float* __restrict__ rel_w_tab,    // [55][64]
    float* __restrict__ relh,               // [B*NH][784][28]
    float* __restrict__ relw)
{
  int gid = blockIdx.x;
  int qi = gid % 784;
  int t2 = gid / 784;
  int h  = t2 % NH;
  int b  = t2 / NH;
  int qy = qi / WOUT, qx = qi % WOUT;
  const _Float16* qrow = qlin + ((size_t)b * NTOK + (qi + 1)) * CCH + h * DHEAD;
  int t = threadIdx.x;
  if (t < 28) {
    const float* R = rel_h_tab + (size_t)(qy - t + 27) * DHEAD;
    float s = 0.f;
#pragma unroll
    for (int c = 0; c < DHEAD; ++c) s += (float)qrow[c] * R[c];
    relh[(((size_t)(b * NH + h)) * 784 + qi) * 28 + t] = s;
  } else if (t < 56) {
    int kx = t - 28;
    const float* R = rel_w_tab + (size_t)(qx - kx + 27) * DHEAD;
    float s = 0.f;
#pragma unroll
    for (int c = 0; c < DHEAD; ++c) s += (float)qrow[c] * R[c];
    relw[(((size_t)(b * NH + h)) * 784 + qi) * 28 + kx] = s;
  }
}

// ---------------- kernel 5: flash attention + bias + residual ----------------
__global__ __launch_bounds__(128) void flash_attn_kernel(
    const _Float16* __restrict__ Q,    // [MROWS][512] (unscaled, post-linear)
    const _Float16* __restrict__ K,
    const _Float16* __restrict__ V,
    const float* __restrict__ RH,      // [B*NH][784][28]
    const float* __restrict__ RW,
    _Float16* __restrict__ O)          // [MROWS][512]
{
  constexpr int LDH = 72;   // 144 B row stride (16B multiple)
  __shared__ alignas(16) _Float16 Qs[64 * LDH];
  __shared__ alignas(16) _Float16 Ks[64 * LDH];
  __shared__ alignas(16) _Float16 VsT[64 * LDH];  // [d][token]
  __shared__ alignas(16) _Float16 Ps[64 * LDH];
  __shared__ float BH[64 * 28], BW[64 * 28];

  const int tid = threadIdx.x, lane = tid & 31, w = tid >> 5;
  const int bh = blockIdx.y, b = bh >> 3, h = bh & 7;
  const int q0 = blockIdx.x * 64;
  const int mrow = w * 16;
  const int cb = lane & 15;            // column within 16-wide tile
  const int ro = (lane >> 4) << 3;     // row offset within 16-row tile
  const _Float16 hscale = (_Float16)0.125f;   // 64^-0.5, exact in f16

  // stage Q tile (pre-scaled) into LDS
  for (int e = tid; e < 64 * 8; e += 128) {
    int r = e >> 3, cseg = (e & 7) << 3;
    int tq = q0 + r;
    v8h q = zero8();
    if (tq < NTOK)
      q = *(const v8h*)&Q[((size_t)b * NTOK + tq) * CCH + h * DHEAD + cseg];
#pragma unroll
    for (int i = 0; i < 8; ++i) q[i] = q[i] * hscale;
    *(v8h*)&Qs[r * LDH + cseg] = q;
  }
  // stage per-row rel bias vectors
  for (int e = tid; e < 64 * 28; e += 128) {
    int r = e / 28, kk = e % 28;
    int tq = q0 + r;
    float vh = 0.f, vw = 0.f;
    if (tq >= 1 && tq < NTOK) {
      size_t rb = (((size_t)bh) * 784 + (tq - 1)) * 28 + kk;
      vh = RH[rb]; vw = RW[rb];
    }
    BH[r * 28 + kk] = vh; BW[r * 28 + kk] = vw;
  }

  float m_i[8], l_i[8];
  v8f oacc[4];
#pragma unroll
  for (int j = 0; j < 8; ++j) { m_i[j] = -1e30f; l_i[j] = 0.f; }
#pragma unroll
  for (int t = 0; t < 4; ++t) oacc[t] = (v8f){0.f,0.f,0.f,0.f,0.f,0.f,0.f,0.f};

  for (int kt = 0; kt < KTILES; ++kt) {
    const int kbase = kt * 64;
    __syncthreads();
#if USE_ASYNC_COPY
    if (kbase + 64 <= NTOK) {
      // K tile: async straight to LDS (overlaps the V transpose below)
      for (int e = tid; e < 64 * 8; e += 128) {
        int r = e >> 3, cseg = (e & 7) << 3;
        size_t o = ((size_t)b * NTOK + kbase + r) * CCH + h * DHEAD + cseg;
        async_cp16(&Ks[r * LDH + cseg], &K[o]);
      }
      for (int e = tid; e < 64 * 8; e += 128) {
        int r = e >> 3, cseg = (e & 7) << 3;
        size_t o = ((size_t)b * NTOK + kbase + r) * CCH + h * DHEAD + cseg;
        v8h vv = *(const v8h*)&V[o];
#pragma unroll
        for (int i = 0; i < 8; ++i) VsT[(cseg + i) * LDH + r] = vv[i];
      }
      async_wait0();
    } else
#endif
    {
      for (int e = tid; e < 64 * 8; e += 128) {
        int r = e >> 3, cseg = (e & 7) << 3;     // r = token, cseg = d segment
        int tk = kbase + r;
        v8h kv = zero8(), vv = zero8();
        if (tk < NTOK) {
          size_t o = ((size_t)b * NTOK + tk) * CCH + h * DHEAD + cseg;
          kv = *(const v8h*)&K[o];
          vv = *(const v8h*)&V[o];
        }
        *(v8h*)&Ks[r * LDH + cseg] = kv;
#pragma unroll
        for (int i = 0; i < 8; ++i) VsT[(cseg + i) * LDH + r] = vv[i];
      }
    }
    if (kt + 1 < KTILES) {  // hint next K tile toward the caches
      int tn = kbase + 64 + (tid >> 1);
      if (tn < NTOK)
        __builtin_prefetch(&K[((size_t)b * NTOK + tn) * CCH + h * DHEAD], 0, 1);
    }
    __syncthreads();

    // S = (Q*scale) @ K^T   (16 x 64 per wave)
    v16h aq0 = frag_a(Qs, LDH, mrow, 0, lane);
    v16h aq1 = frag_a(Qs, LDH, mrow, 32, lane);
    v8f s[4];
#pragma unroll
    for (int nt = 0; nt < 4; ++nt) {
      v8f z = (v8f){0.f,0.f,0.f,0.f,0.f,0.f,0.f,0.f};
      z = wmma_f16(aq0, frag_bT(Ks, LDH, nt * 16, 0, lane), z);
      z = wmma_f16(aq1, frag_bT(Ks, LDH, nt * 16, 32, lane), z);
      s[nt] = z;
    }

    // bias + masking
#pragma unroll
    for (int nt = 0; nt < 4; ++nt) {
      int tk = kbase + nt * 16 + cb;
      int ki = (tk >= 1 && tk < NTOK) ? (tk - 1) : 0;
      int ky = ki / WOUT, kx = ki % WOUT;
#pragma unroll
      for (int j = 0; j < 8; ++j) {
        int rloc = mrow + ro + j;
        int tq = q0 + rloc;
        float sv = s[nt][j];
        if (tk >= NTOK || tq >= NTOK) sv = -1e30f;
        else if (tk >= 1 && tq >= 1)  sv += BH[rloc * 28 + ky] + BW[rloc * 28 + kx];
        s[nt][j] = sv;
      }
    }

    // online softmax (row reductions across 16-lane half groups)
    float sc[8];
#pragma unroll
    for (int j = 0; j < 8; ++j) {
      float mx = fmaxf(fmaxf(s[0][j], s[1][j]), fmaxf(s[2][j], s[3][j]));
#pragma unroll
      for (int off = 1; off < 16; off <<= 1) mx = fmaxf(mx, __shfl_xor(mx, off, 32));
      float mnew = fmaxf(m_i[j], mx);
      sc[j] = __expf(m_i[j] - mnew);
      float rsum = 0.f;
#pragma unroll
      for (int nt = 0; nt < 4; ++nt) {
        float p = __expf(s[nt][j] - mnew);
        s[nt][j] = p; rsum += p;
      }
#pragma unroll
      for (int off = 1; off < 16; off <<= 1) rsum += __shfl_xor(rsum, off, 32);
      l_i[j] = l_i[j] * sc[j] + rsum;
      m_i[j] = mnew;
    }
#pragma unroll
    for (int t = 0; t < 4; ++t)
#pragma unroll
      for (int j = 0; j < 8; ++j) oacc[t][j] *= sc[j];

    // write P (D-layout -> row-major LDS) for the second WMMA
#pragma unroll
    for (int nt = 0; nt < 4; ++nt)
#pragma unroll
      for (int j = 0; j < 8; ++j)
        Ps[(size_t)(mrow + ro + j) * LDH + nt * 16 + cb] = (_Float16)s[nt][j];
    __syncthreads();

    // O += P @ V   (B[k][n] = V[k][n] = VsT[n][k] -> contiguous frag_bT)
    v16h ap0 = frag_a(Ps, LDH, mrow, 0, lane);
    v16h ap1 = frag_a(Ps, LDH, mrow, 32, lane);
#pragma unroll
    for (int nt = 0; nt < 4; ++nt) {
      oacc[nt] = wmma_f16(ap0, frag_bT(VsT, LDH, nt * 16, 0, lane), oacc[nt]);
      oacc[nt] = wmma_f16(ap1, frag_bT(VsT, LDH, nt * 16, 32, lane), oacc[nt]);
    }
  }

  // epilogue: 1/l, residual (+q for non-cls), store f16
#pragma unroll
  for (int nt = 0; nt < 4; ++nt) {
    int col = nt * 16 + cb;
#pragma unroll
    for (int j = 0; j < 8; ++j) {
      int tq = q0 + mrow + ro + j;
      if (tq < NTOK) {
        float inv = (l_i[j] > 0.f) ? (1.f / l_i[j]) : 0.f;
        float v = oacc[nt][j] * inv;
        size_t o = ((size_t)b * NTOK + tq) * CCH + h * DHEAD + col;
        if (tq >= 1) v += (float)Q[o];
        O[o] = (_Float16)v;
      }
    }
  }
}

// ---------------- host-side launch ----------------
extern "C" void kernel_launch(void* const* d_in, const int* in_sizes, int n_in,
                              void* d_out, int out_size, void* d_ws, size_t ws_size,
                              hipStream_t stream) {
  (void)in_sizes; (void)n_in; (void)out_size; (void)ws_size;
  const float* x     = (const float*)d_in[0];
  const float* x_ref = (const float*)d_in[1];
  const float* w_q   = (const float*)d_in[2];
  const float* b_q   = (const float*)d_in[3];
  const float* w_k   = (const float*)d_in[4];
  const float* b_k   = (const float*)d_in[5];
  const float* w_v   = (const float*)d_in[6];
  const float* b_v   = (const float*)d_in[7];
  const float* w_p   = (const float*)d_in[8];
  const float* b_p   = (const float*)d_in[9];
  const float* cw_q  = (const float*)d_in[10];
  const float* cw_k  = (const float*)d_in[11];
  const float* cw_v  = (const float*)d_in[12];
  const float* g_q   = (const float*)d_in[13];
  const float* bt_q  = (const float*)d_in[14];
  const float* g_k   = (const float*)d_in[15];
  const float* bt_k  = (const float*)d_in[16];
  const float* g_v   = (const float*)d_in[17];
  const float* bt_v  = (const float*)d_in[18];
  const float* rph   = (const float*)d_in[19];
  const float* rpw   = (const float*)d_in[20];
  float* outf = (float*)d_out;

  char* ws = (char*)d_ws;
  size_t off = 0;
  auto alloc = [&](size_t bytes) -> void* {
    void* p = ws + off;
    off += (bytes + 255) & ~(size_t)255;
    return p;
  };
  const size_t WBYTES = (size_t)512 * 512 * sizeof(_Float16);
  const size_t ABYTES = (size_t)MROWS * 512 * sizeof(_Float16);
  _Float16* wqh   = (_Float16*)alloc(WBYTES);
  _Float16* wkh   = (_Float16*)alloc(WBYTES);
  _Float16* wvh   = (_Float16*)alloc(WBYTES);
  _Float16* wph   = (_Float16*)alloc(WBYTES);
  _Float16* qpool = (_Float16*)alloc(ABYTES);
  _Float16* kpool = (_Float16*)alloc(ABYTES);
  _Float16* vpool = (_Float16*)alloc(ABYTES);
  _Float16* qlin  = (_Float16*)alloc(ABYTES);
  _Float16* klin  = (_Float16*)alloc(ABYTES);
  _Float16* vlin  = (_Float16*)alloc(ABYTES);
  _Float16* aout  = (_Float16*)alloc(ABYTES);
  float* relh = (float*)alloc((size_t)BATCH * NH * 784 * 28 * sizeof(float));
  float* relw = (float*)alloc((size_t)BATCH * NH * 784 * 28 * sizeof(float));

  // 1) weights to f16
  {
    int n = 512 * 512, blocks = (n + 255) / 256;
    cvt_f32_f16<<<blocks, 256, 0, stream>>>(w_q, wqh, n);
    cvt_f32_f16<<<blocks, 256, 0, stream>>>(w_k, wkh, n);
    cvt_f32_f16<<<blocks, 256, 0, stream>>>(w_v, wvh, n);
    cvt_f32_f16<<<blocks, 256, 0, stream>>>(w_p, wph, n);
  }

  // 2) pool + LN (q from x, k/v from x_ref)
  {
    int blocks = BATCH * NH * NTOK;
    pool_ln_kernel<<<blocks, 64, 0, stream>>>(x,     cw_q, g_q, bt_q, qpool);
    pool_ln_kernel<<<blocks, 64, 0, stream>>>(x_ref, cw_k, g_k, bt_k, kpool);
    pool_ln_kernel<<<blocks, 64, 0, stream>>>(x_ref, cw_v, g_v, bt_v, vpool);
  }

  // 3) q/k/v linear projections (WMMA)
  {
    dim3 grid((MROWS + 127) / 128, 512 / 64);
    gemm512_wmma<0><<<grid, 256, 0, stream>>>(qpool, wqh, b_q, qlin, nullptr, MROWS);
    gemm512_wmma<0><<<grid, 256, 0, stream>>>(kpool, wkh, b_k, klin, nullptr, MROWS);
    gemm512_wmma<0><<<grid, 256, 0, stream>>>(vpool, wvh, b_v, vlin, nullptr, MROWS);
  }

  // 4) rel-pos projections
  relpos_kernel<<<BATCH * NH * 784, 64, 0, stream>>>(qlin, rph, rpw, relh, relw);

  // 5) flash attention (+bias, +residual)
  {
    dim3 grid(KTILES, BATCH * NH);
    flash_attn_kernel<<<grid, 128, 0, stream>>>(qlin, klin, vlin, relh, relw, aout);
  }

  // 6) final projection -> f32 output
  {
    dim3 grid((MROWS + 127) / 128, 512 / 64);
    gemm512_wmma<1><<<grid, 256, 0, stream>>>(aout, wph, b_p, nullptr, outf, MROWS);
  }
}
